// PointMLP_86268713107580
// MI455X (gfx1250) — matrix operand order repaired and checked by reading, not compiled, not run on hardware
//
#include <hip/hip_runtime.h>
#include <hip/hip_bf16.h>
#include <math.h>

typedef __attribute__((ext_vector_type(16))) _Float16 v16h;
typedef __attribute__((ext_vector_type(8)))  _Float16 v8h;
typedef __attribute__((ext_vector_type(8)))  float    v8f;
typedef _Float16 half_t;

#define BN_EPS 1e-5f

// ---------------------------------------------------------------------------
// Param prep: fold BN (+optional linear bias) into per-channel scale/bias
// ---------------------------------------------------------------------------
__global__ void pm_fuse_bn(const float* __restrict__ g, const float* __restrict__ b,
                           const float* __restrict__ m, const float* __restrict__ v,
                           const float* __restrict__ extra_bias,
                           float* __restrict__ scale, float* __restrict__ bias, int C) {
  int i = blockIdx.x * blockDim.x + threadIdx.x;
  if (i >= C) return;
  float s = g[i] / sqrtf(v[i] + BN_EPS);
  float eb = extra_bias ? extra_bias[i] : 0.0f;
  scale[i] = s;
  bias[i]  = (eb - m[i]) * s + b[i];
}

// Convert f32 weight (OC x K) into WMMA-B-fragment-native f16 layout:
//   Wh[kc][tn][lane][j]  with kc = k/32, tn = n/16, n%16 = lane%16, k%32 = 8*(lane/16) + (j%8) + 16*(j/8)
// so one lane's 16 halves for a (k-chunk, n-tile) are contiguous (32B).
__global__ void pm_convert_w(const float* __restrict__ W, half_t* __restrict__ Wh,
                             int OC, int K, int Kp) {
  int i = blockIdx.x * blockDim.x + threadIdx.x;
  if (i >= OC * Kp) return;
  int j    = i & 15;                 // fragment element (k slot)
  int lane = (i >> 4) & 31;          // fragment lane (n = lane%16, k half = lane/16)
  int rest = i >> 9;
  int ntn  = OC >> 4;
  int tn   = rest % ntn;
  int kc   = rest / ntn;
  int n = tn * 16 + (lane & 15);
  int k = kc * 32 + ((lane >> 4) << 3) + (j & 7) + ((j >> 3) << 4);
  Wh[i] = (half_t)((k < K) ? W[(size_t)n * K + k] : 0.0f);
}

// ---------------------------------------------------------------------------
// Embed input pack: x (B*N, 9) f32 -> A (B*N, 32) f16 (padded); also xyz f32
// ---------------------------------------------------------------------------
__global__ void pm_pack_embed(const float* __restrict__ x, half_t* __restrict__ A,
                              float* __restrict__ xyz, int BN) {
  int i = blockIdx.x * blockDim.x + threadIdx.x;
  if (i >= BN * 32) return;
  int ch = i & 31;
  int r  = i >> 5;
  float v = (ch < 9) ? x[(size_t)r * 9 + ch] : 0.0f;
  A[i] = (half_t)v;
  if (ch < 3) xyz[(size_t)r * 3 + ch] = v;
}

// ---------------------------------------------------------------------------
// Fused WMMA GEMM: out = act( (A @ W^T) * scale + bias [+ resid] )
//   A: M x Kp (f16, row-major), W: fragment-native (see pm_convert_w)
//   one wave32 computes one 16x16 tile via v_wmma_f32_16x16x32_f16
//   tiles ordered column-major so the 4 waves of a block share B fragments
//   K-loop software-pipelined: chunk kc+1 loads issue before WMMA on kc
// ---------------------------------------------------------------------------
__global__ __launch_bounds__(128)
void pm_gemm_bn(const half_t* __restrict__ A, const half_t* __restrict__ W,
                half_t* __restrict__ out,
                const float* __restrict__ scale, const float* __restrict__ bias,
                const half_t* __restrict__ resid,
                int M, int N, int Kp, int do_relu) {
  const int lane  = threadIdx.x & 31;
  const int wave  = blockIdx.x * (blockDim.x >> 5) + (threadIdx.x >> 5);
  const int ntm   = M >> 4;
  const int ntn   = N >> 4;
  if (wave >= ntm * ntn) return;
  const int tn = wave / ntm;            // column-major: block-mates share tn
  const int tm = wave - tn * ntm;
  const int m0 = tm << 4, n0 = tn << 4;

  v8f acc = {};
  const int row  = lane & 15;
  const int koff = (lane < 16) ? 0 : 8;                 // A layout per ISA 7.12.2
  const half_t* arow  = A + (size_t)(m0 + row) * Kp + koff;
  const v16h*   wfrag = (const v16h*)W + ((size_t)tn << 5) + lane;
  const size_t  wstep = (size_t)ntn << 5;               // v16h per k-chunk

  const int nk = Kp >> 5;

  // prologue: chunk 0
  v8h alo = *(const v8h*)(arow);
  v8h ahi = *(const v8h*)(arow + 16);
  v16h b  = *wfrag;

  for (int kc = 0; kc + 1 < nk; ++kc) {
    // issue next-chunk loads (and a distance-2 speculative prefetch) first,
    // then WMMA on the already-resident fragments -> loads hide behind MMA
    const half_t* anext = arow + ((kc + 1) << 5);
    __builtin_prefetch(anext + 32, 0, 1);
    __builtin_prefetch((const half_t*)(wfrag + 2 * wstep), 0, 1);
    v8h nlo = *(const v8h*)(anext);
    v8h nhi = *(const v8h*)(anext + 16);
    v16h nb = *(wfrag + wstep);
    v16h a  = __builtin_shufflevector(alo, ahi, 0, 1, 2, 3, 4, 5, 6, 7,
                                      8, 9, 10, 11, 12, 13, 14, 15);
    acc = __builtin_amdgcn_wmma_f32_16x16x32_f16(false, a, false, b,
                                                 (short)0, acc, false, false);
    alo = nlo; ahi = nhi; b = nb;
    wfrag += wstep;
  }
  // epilogue: last chunk
  {
    v16h a = __builtin_shufflevector(alo, ahi, 0, 1, 2, 3, 4, 5, 6, 7,
                                     8, 9, 10, 11, 12, 13, 14, 15);
    acc = __builtin_amdgcn_wmma_f32_16x16x32_f16(false, a, false, b,
                                                 (short)0, acc, false, false);
  }

  const int col = n0 + (lane & 15);
  const float s  = scale ? scale[col] : 1.0f;
  const float bi = bias  ? bias[col]  : 0.0f;
#pragma unroll
  for (int r = 0; r < 8; ++r) {                         // C/D layout per ISA 7.12.2
    int mrow = m0 + ((lane < 16) ? r : r + 8);
    float v = acc[r] * s + bi;
    size_t o = (size_t)mrow * N + col;
    if (resid)   v += (float)resid[o];
    if (do_relu) v = v > 0.0f ? v : 0.0f;
    out[o] = (half_t)v;
  }
}

// ---------------------------------------------------------------------------
// Farthest point sampling: one block per batch, sequential scan like reference
// ---------------------------------------------------------------------------
__global__ void pm_fps(const float* __restrict__ xyz, int N, int S,
                       int* __restrict__ fps_idx) {
  int b = blockIdx.x;
  int t = threadIdx.x;                       // 256 threads
  const float* p = xyz + (size_t)b * N * 3;
  __shared__ float dists[512];
  __shared__ float red[256];
  __shared__ int   redi[256];
  __shared__ int   s_last;
  for (int i = t; i < N; i += 256) dists[i] = 1e10f;
  if (t == 0) { s_last = 0; fps_idx[(size_t)b * S] = 0; }
  __syncthreads();
  for (int s = 1; s < S; ++s) {
    int last = s_last;
    float lx = p[last * 3], ly = p[last * 3 + 1], lz = p[last * 3 + 2];
    float best = -1.0f; int besti = 0;
    for (int i = t; i < N; i += 256) {
      float dx = p[i * 3] - lx, dy = p[i * 3 + 1] - ly, dz = p[i * 3 + 2] - lz;
      float d  = dx * dx + dy * dy + dz * dz;
      float dm = fminf(dists[i], d);
      dists[i] = dm;
      if (dm > best) { best = dm; besti = i; }
    }
    red[t] = best; redi[t] = besti;
    __syncthreads();
    for (int off = 128; off > 0; off >>= 1) {
      if (t < off) {
        if (red[t + off] > red[t] ||
            (red[t + off] == red[t] && redi[t + off] < redi[t])) {
          red[t] = red[t + off]; redi[t] = redi[t + off];
        }
      }
      __syncthreads();
    }
    if (t == 0) { s_last = redi[0]; fps_idx[(size_t)b * S + s] = redi[0]; }
    __syncthreads();
  }
}

// ---------------------------------------------------------------------------
// KNN (k=12): one thread per (b, center); also writes gathered new_xyz
// ---------------------------------------------------------------------------
__global__ void pm_knn(const float* __restrict__ xyz, const int* __restrict__ fps_idx,
                       float* __restrict__ new_xyz, int* __restrict__ knn_idx,
                       int B, int N, int S) {
  int id = blockIdx.x * blockDim.x + threadIdx.x;
  if (id >= B * S) return;
  int b = id / S;
  const float* p = xyz + (size_t)b * N * 3;
  int ci = fps_idx[id];
  float cx = p[ci * 3], cy = p[ci * 3 + 1], cz = p[ci * 3 + 2];
  new_xyz[(size_t)id * 3]     = cx;
  new_xyz[(size_t)id * 3 + 1] = cy;
  new_xyz[(size_t)id * 3 + 2] = cz;
  float bd[12]; int bi[12];
#pragma unroll
  for (int j = 0; j < 12; ++j) { bd[j] = 1e30f; bi[j] = 0; }
  for (int i = 0; i < N; ++i) {
    float dx = p[i * 3] - cx, dy = p[i * 3 + 1] - cy, dz = p[i * 3 + 2] - cz;
    float d  = dx * dx + dy * dy + dz * dz;
    if (d < bd[11]) {
      int j = 11;
      while (j > 0 && bd[j - 1] > d) { bd[j] = bd[j - 1]; bi[j] = bi[j - 1]; --j; }
      bd[j] = d; bi[j] = i;
    }
  }
#pragma unroll
  for (int j = 0; j < 12; ++j) knn_idx[(size_t)id * 12 + j] = bi[j];
}

// ---------------------------------------------------------------------------
// Per-batch sum/sumsq of (grouped - anchor) for std (one block per batch)
// ---------------------------------------------------------------------------
__global__ void pm_stats(const half_t* __restrict__ pts, const float* __restrict__ xyz,
                         const float* __restrict__ new_xyz,
                         const int* __restrict__ fps_idx, const int* __restrict__ knn_idx,
                         float* __restrict__ stats, int Np, int S, int C) {
  int b = blockIdx.x, t = threadIdx.x;
  int D = C + 3;
  long long total = (long long)S * 12 * D;
  float sum = 0.0f, sq = 0.0f;
  for (long long e = t; e < total; e += blockDim.x) {
    int ch = (int)(e % D);
    long long r = e / D;
    int k = (int)(r % 12);
    int s = (int)(r / 12);
    int center = fps_idx[(size_t)b * S + s];
    int nb     = knn_idx[((size_t)b * S + s) * 12 + k];
    float g, a;
    if (ch < C) {
      g = (float)pts[((size_t)b * Np + nb) * C + ch];
      a = (float)pts[((size_t)b * Np + center) * C + ch];
    } else {
      g = xyz[((size_t)b * Np + nb) * 3 + (ch - C)];
      a = new_xyz[((size_t)b * S + s) * 3 + (ch - C)];
    }
    float d = g - a;
    sum += d; sq += d * d;
  }
  __shared__ float ss[256], s2[256];
  ss[t] = sum; s2[t] = sq;
  __syncthreads();
  for (int off = 128; off > 0; off >>= 1) {
    if (t < off) { ss[t] += ss[t + off]; s2[t] += s2[t + off]; }
    __syncthreads();
  }
  if (t == 0) { stats[b * 2] = ss[0]; stats[b * 2 + 1] = s2[0]; }
}

// ---------------------------------------------------------------------------
// Build grouped A-matrix rows (B*S*12, Kp): [alpha*(g-anchor)/std+beta | rep]
// ---------------------------------------------------------------------------
__global__ void pm_pack_group(const half_t* __restrict__ pts, const float* __restrict__ xyz,
                              const float* __restrict__ new_xyz,
                              const int* __restrict__ fps_idx, const int* __restrict__ knn_idx,
                              const float* __restrict__ stats,
                              const float* __restrict__ alpha, const float* __restrict__ beta,
                              half_t* __restrict__ out, int Np, int S, int C, int Kp, int B) {
  size_t id = (size_t)blockIdx.x * blockDim.x + threadIdx.x;
  int D = C + 3;
  size_t rows = (size_t)B * S * 12;
  if (id >= rows * Kp) return;
  int ch = (int)(id % Kp);
  size_t r = id / Kp;
  int k = (int)(r % 12);
  size_t rs = r / 12;
  int s = (int)(rs % S);
  int b = (int)(rs / S);
  float v = 0.0f;
  if (ch < 2 * C + 3) {
    int center = fps_idx[(size_t)b * S + s];
    if (ch < D) {
      int nb = knn_idx[((size_t)b * S + s) * 12 + k];
      float g, a;
      if (ch < C) {
        g = (float)pts[((size_t)b * Np + nb) * C + ch];
        a = (float)pts[((size_t)b * Np + center) * C + ch];
      } else {
        g = xyz[((size_t)b * Np + nb) * 3 + (ch - C)];
        a = new_xyz[((size_t)b * S + s) * 3 + (ch - C)];
      }
      float n    = (float)S * 12.0f * (float)D;
      float mean = stats[b * 2] / n;
      float var  = (stats[b * 2 + 1] - n * mean * mean) / (n - 1.0f);
      float sd   = sqrtf(fmaxf(var, 0.0f));
      v = alpha[ch] * ((g - a) / (sd + 1e-5f)) + beta[ch];
    } else {
      v = (float)pts[((size_t)b * Np + center) * C + (ch - D)];   // rep = new_pts
    }
  }
  out[id] = (half_t)v;
}

// max over the k=12 group rows: y (rows*12, C) -> pts (rows, C)
__global__ void pm_maxpool(const half_t* __restrict__ y, half_t* __restrict__ pts,
                           size_t rows, int C) {
  size_t i = (size_t)blockIdx.x * blockDim.x + threadIdx.x;
  if (i >= rows * C) return;
  int ch = (int)(i % C);
  size_t r = i / C;
  const half_t* p = y + (r * 12) * (size_t)C + ch;
  float m = -1e30f;
#pragma unroll
  for (int k = 0; k < 12; ++k) m = fmaxf(m, (float)p[(size_t)k * C]);
  pts[i] = (half_t)m;
}

// global max over remaining points: pts (B*S, C) -> g (B, C)
__global__ void pm_headpool(const half_t* __restrict__ pts, half_t* __restrict__ g,
                            int S, int C) {
  int i = blockIdx.x * blockDim.x + threadIdx.x;
  if (i >= 128 * C) return;
  int ch = i % C, b = i / C;
  float m = -1e30f;
  for (int n = 0; n < S; ++n)
    m = fmaxf(m, (float)pts[((size_t)(b * S + n)) * C + ch]);
  g[i] = (half_t)m;
}

// final 6-class projection, f32 output
__global__ void pm_head_final(const half_t* __restrict__ h2, const float* __restrict__ W3,
                              const float* __restrict__ b3, float* __restrict__ out) {
  int i = blockIdx.x * blockDim.x + threadIdx.x;
  if (i >= 128 * 6) return;
  int j = i % 6, b = i / 6;
  float s = b3[j];
  for (int c = 0; c < 256; ++c)
    s += (float)h2[(size_t)b * 256 + c] * W3[(size_t)j * 256 + c];
  out[i] = s;
}

// ---------------------------------------------------------------------------
extern "C" void kernel_launch(void* const* d_in, const int* in_sizes, int n_in,
                              void* d_out, int out_size, void* d_ws, size_t ws_size,
                              hipStream_t stream) {
  (void)in_sizes; (void)n_in; (void)out_size; (void)ws_size;
  const int B = 128;
  const int STAGE_C[4] = {64, 128, 256, 512};
  const int OUTC[4]    = {128, 256, 512, 1024};
  const int GRP[4]     = {256, 128, 64, 32};
  const int KPT[4]     = {160, 288, 544, 1056};   // roundup32(3 + 2c)
  const int NPTS[4]    = {512, 256, 128, 64};

  const float* x = (const float*)d_in[0];

  // ---- workspace arena -------------------------------------------------
  char* base = (char*)d_ws;
  size_t off = 0;
  auto alloc = [&](size_t bytes) -> void* {
    void* p = base + off;
    off += (bytes + 255) & ~(size_t)255;
    return p;
  };
  auto cdiv = [](size_t a, size_t b) { return (int)((a + b - 1) / b); };

  const size_t ACT = (size_t)B * 512 * 64;                     // 4.19M halves
  half_t* ptsA = (half_t*)alloc(sizeof(half_t) * ACT);
  half_t* ptsB = (half_t*)alloc(sizeof(half_t) * ACT);
  half_t* tmpP = (half_t*)alloc(sizeof(half_t) * ACT);
  float*  xyz0  = (float*)alloc(sizeof(float) * (size_t)B * 512 * 3);
  float*  nxyzA = (float*)alloc(sizeof(float) * (size_t)B * 256 * 3);
  float*  nxyzB = (float*)alloc(sizeof(float) * (size_t)B * 256 * 3);
  int*    fpsI  = (int*)alloc(sizeof(int) * (size_t)B * 256);
  int*    knnI  = (int*)alloc(sizeof(int) * (size_t)B * 256 * 12);
  float*  stats = (float*)alloc(sizeof(float) * (size_t)B * 2);
  half_t* grouped = (half_t*)alloc(sizeof(half_t) * (size_t)B * 256 * 12 * 160); // 62.9M, also res tmp
  half_t* ybuf    = (half_t*)alloc(sizeof(half_t) * (size_t)B * 256 * 12 * 128); // 50.3M
  half_t* embA    = (half_t*)alloc(sizeof(half_t) * (size_t)B * 512 * 32);
  half_t* gbuf    = (half_t*)alloc(sizeof(half_t) * 128 * 1024);
  half_t* h1      = (half_t*)alloc(sizeof(half_t) * 128 * 512);
  half_t* h2      = (half_t*)alloc(sizeof(half_t) * 128 * 256);

  // ---- weight / BN prep -------------------------------------------------
  half_t* embW = (half_t*)alloc(sizeof(half_t) * 64 * 32);
  float*  embS = (float*)alloc(sizeof(float) * 64);
  float*  embB = (float*)alloc(sizeof(float) * 64);
  half_t* tW[4]; float* tS[4]; float* tB[4];
  half_t* rW[4][8]; float* rS[4][8]; float* rB[4][8];
  for (int i = 0; i < 4; ++i) {
    tW[i] = (half_t*)alloc(sizeof(half_t) * (size_t)OUTC[i] * KPT[i]);
    tS[i] = (float*)alloc(sizeof(float) * OUTC[i]);
    tB[i] = (float*)alloc(sizeof(float) * OUTC[i]);
    for (int j = 0; j < 8; ++j) {
      rW[i][j] = (half_t*)alloc(sizeof(half_t) * (size_t)OUTC[i] * OUTC[i]);
      rS[i][j] = (float*)alloc(sizeof(float) * OUTC[i]);
      rB[i][j] = (float*)alloc(sizeof(float) * OUTC[i]);
    }
  }
  half_t* hW1 = (half_t*)alloc(sizeof(half_t) * 512 * 1024);
  float*  hS1 = (float*)alloc(sizeof(float) * 512);
  float*  hB1 = (float*)alloc(sizeof(float) * 512);
  half_t* hW2 = (half_t*)alloc(sizeof(half_t) * 256 * 512);
  float*  hS2 = (float*)alloc(sizeof(float) * 256);
  float*  hB2 = (float*)alloc(sizeof(float) * 256);

  auto F = [&](int idx) { return (const float*)d_in[idx]; };

  // embed: leaves 1..5
  pm_convert_w<<<cdiv(64 * 32, 256), 256, 0, stream>>>(F(1), embW, 64, 9, 32);
  pm_fuse_bn<<<1, 64, 0, stream>>>(F(2), F(3), F(4), F(5), nullptr, embS, embB, 64);

  for (int i = 0; i < 4; ++i) {
    int pb = 6 + i * 47;
    int oc = OUTC[i], c = STAGE_C[i], d = 3 + 2 * c;
    pm_convert_w<<<cdiv((size_t)oc * KPT[i], 256), 256, 0, stream>>>(F(pb + 2), tW[i], oc, d, KPT[i]);
    pm_fuse_bn<<<cdiv(oc, 256), 256, 0, stream>>>(F(pb + 3), F(pb + 4), F(pb + 5), F(pb + 6),
                                                  nullptr, tS[i], tB[i], oc);
    for (int rb = 0; rb < 4; ++rb) {                 // pre0, pre1, pos0, pos1
      int wb = pb + 7 + rb * 10;
      pm_convert_w<<<cdiv((size_t)oc * oc, 256), 256, 0, stream>>>(F(wb), rW[i][rb * 2], oc, oc, oc);
      pm_fuse_bn<<<cdiv(oc, 256), 256, 0, stream>>>(F(wb + 1), F(wb + 2), F(wb + 3), F(wb + 4),
                                                    nullptr, rS[i][rb * 2], rB[i][rb * 2], oc);
      pm_convert_w<<<cdiv((size_t)oc * oc, 256), 256, 0, stream>>>(F(wb + 5), rW[i][rb * 2 + 1], oc, oc, oc);
      pm_fuse_bn<<<cdiv(oc, 256), 256, 0, stream>>>(F(wb + 6), F(wb + 7), F(wb + 8), F(wb + 9),
                                                    nullptr, rS[i][rb * 2 + 1], rB[i][rb * 2 + 1], oc);
    }
  }
  // head: 194..207 (biases folded into BN bias)
  pm_convert_w<<<cdiv(512 * 1024, 256), 256, 0, stream>>>(F(194), hW1, 512, 1024, 1024);
  pm_fuse_bn<<<cdiv(512, 256), 256, 0, stream>>>(F(196), F(197), F(198), F(199), F(195), hS1, hB1, 512);
  pm_convert_w<<<cdiv(256 * 512, 256), 256, 0, stream>>>(F(200), hW2, 256, 512, 512);
  pm_fuse_bn<<<1, 256, 0, stream>>>(F(202), F(203), F(204), F(205), F(201), hS2, hB2, 256);

  auto gemm = [&](const half_t* A, const half_t* W, half_t* out, const float* sc,
                  const float* bi, const half_t* resid, size_t M, int N, int Kp, int relu) {
    size_t tiles = (M >> 4) * (size_t)(N >> 4);
    pm_gemm_bn<<<cdiv(tiles, 4), 128, 0, stream>>>(A, W, out, sc, bi, resid,
                                                   (int)M, N, Kp, relu);
  };

  // ---- forward ----------------------------------------------------------
  pm_pack_embed<<<cdiv((size_t)B * 512 * 32, 256), 256, 0, stream>>>(x, embA, xyz0, B * 512);
  gemm(embA, embW, ptsA, embS, embB, nullptr, (size_t)B * 512, 64, 32, 1);

  half_t* pts = ptsA;
  half_t* ptsNext = ptsB;
  float* xyzCur = xyz0;
  float* nx = nxyzA;
  float* nxOther = nxyzB;

  for (int i = 0; i < 4; ++i) {
    int pb = 6 + i * 47;
    int S = GRP[i], Np = NPTS[i], c = STAGE_C[i], oc = OUTC[i], Kp = KPT[i];
    size_t rows = (size_t)B * S * 12;

    pm_fps<<<B, 256, 0, stream>>>(xyzCur, Np, S, fpsI);
    pm_knn<<<cdiv((size_t)B * S, 128), 128, 0, stream>>>(xyzCur, fpsI, nx, knnI, B, Np, S);
    pm_stats<<<B, 256, 0, stream>>>(pts, xyzCur, nx, fpsI, knnI, stats, Np, S, c);
    pm_pack_group<<<cdiv(rows * Kp, 256), 256, 0, stream>>>(pts, xyzCur, nx, fpsI, knnI, stats,
                                                            F(pb), F(pb + 1), grouped,
                                                            Np, S, c, Kp, B);
    // transfer conv + BN + relu
    gemm(grouped, tW[i], ybuf, tS[i], tB[i], nullptr, rows, oc, Kp, 1);
    // pre residual blocks (tmp aliases `grouped`, safe in-place residual write)
    for (int rb = 0; rb < 2; ++rb) {
      gemm(ybuf, rW[i][rb * 2], grouped, rS[i][rb * 2], rB[i][rb * 2], nullptr, rows, oc, oc, 1);
      gemm(grouped, rW[i][rb * 2 + 1], ybuf, rS[i][rb * 2 + 1], rB[i][rb * 2 + 1], ybuf, rows, oc, oc, 1);
    }
    // max over the k=12 neighbors
    pm_maxpool<<<cdiv((size_t)B * S * oc, 256), 256, 0, stream>>>(ybuf, ptsNext, (size_t)B * S, oc);
    // pos residual blocks
    for (int rb = 2; rb < 4; ++rb) {
      gemm(ptsNext, rW[i][rb * 2], tmpP, rS[i][rb * 2], rB[i][rb * 2], nullptr, (size_t)B * S, oc, oc, 1);
      gemm(tmpP, rW[i][rb * 2 + 1], ptsNext, rS[i][rb * 2 + 1], rB[i][rb * 2 + 1], ptsNext,
           (size_t)B * S, oc, oc, 1);
    }
    pts = ptsNext;
    ptsNext = (pts == ptsA) ? ptsB : ptsA;
    xyzCur = nx;
    { float* t = nx; nx = nxOther; nxOther = t; }
  }

  // ---- head -------------------------------------------------------------
  pm_headpool<<<cdiv(128 * 1024, 256), 256, 0, stream>>>(pts, gbuf, 32, 1024);
  gemm(gbuf, hW1, h1, hS1, hB1, nullptr, 128, 512, 1024, 1);
  gemm(h1, hW2, h2, hS2, hB2, nullptr, 128, 256, 512, 1);
  pm_head_final<<<cdiv(128 * 6, 128), 128, 0, stream>>>(h2, F(206), F(207), (float*)d_out);
}
